// SimpleSNN_12575664242696
// MI455X (gfx1250) — matrix operand order repaired and checked by reading, not compiled, not run on hardware
//
#include <hip/hip_runtime.h>

typedef __attribute__((ext_vector_type(16))) _Float16 v16h;
typedef __attribute__((ext_vector_type(8)))  _Float16 v8h;
typedef __attribute__((ext_vector_type(8)))  float    v8f;

static constexpr int BSZ  = 4096;  // batch
static constexpr int INF  = 1024;  // fc1 in-features
static constexpr int HDIM = 2048;  // hidden
static constexpr int ODIM = 256;   // output

// ===========================================================================
// f16 fragment loaders (fast path): data already converted, pure b128 loads.
// A fragment (16x32 MxK): lane holds row (lane&15); two 8-half chunks at
//   kb = k0 + (lane>>4)*8 and kb+16.
// B fragment (32x16 KxN): lane holds col (lane&15); two 8-half chunks at
//   kb = k0 + (lane>>4)*8 and kb+16.
// ===========================================================================
__device__ __forceinline__ v16h load_fragA_h(const _Float16* __restrict__ row, int kb) {
  v8h lo = *reinterpret_cast<const v8h*>(row + kb);
  v8h hi = *reinterpret_cast<const v8h*>(row + kb + 16);
  return __builtin_shufflevector(lo, hi, 0,1,2,3,4,5,6,7,8,9,10,11,12,13,14,15);
}
__device__ __forceinline__ v16h load_fragB_h(const _Float16* __restrict__ row, int kb) {
  v8h lo = *reinterpret_cast<const v8h*>(row + kb);
  v8h hi = *reinterpret_cast<const v8h*>(row + kb + 16);
  return __builtin_shufflevector(lo, hi, 0,1,2,3,4,5,6,7,8,9,10,11,12,13,14,15);
}

// ---- f32 fragment loaders (fallback path, inline conversion) ----
__device__ __forceinline__ v16h load_fragA(const float* __restrict__ row, int kb) {
  const float4* p0 = reinterpret_cast<const float4*>(row + kb);
  const float4* p1 = reinterpret_cast<const float4*>(row + kb + 16);
  float4 a0 = p0[0], a1 = p0[1], c0 = p1[0], c1 = p1[1];
  v16h r;
  r[0]  = (_Float16)a0.x; r[1]  = (_Float16)a0.y; r[2]  = (_Float16)a0.z; r[3]  = (_Float16)a0.w;
  r[4]  = (_Float16)a1.x; r[5]  = (_Float16)a1.y; r[6]  = (_Float16)a1.z; r[7]  = (_Float16)a1.w;
  r[8]  = (_Float16)c0.x; r[9]  = (_Float16)c0.y; r[10] = (_Float16)c0.z; r[11] = (_Float16)c0.w;
  r[12] = (_Float16)c1.x; r[13] = (_Float16)c1.y; r[14] = (_Float16)c1.z; r[15] = (_Float16)c1.w;
  return r;
}
__device__ __forceinline__ v16h load_fragB(const float* __restrict__ row, int kb) {
  const float4* p = reinterpret_cast<const float4*>(row + kb);
  const float4* q = reinterpret_cast<const float4*>(row + kb + 16);
  float4 a0 = p[0], a1 = p[1], a2 = q[0], a3 = q[1];
  v16h r;
  r[0]  = (_Float16)a0.x; r[1]  = (_Float16)a0.y; r[2]  = (_Float16)a0.z; r[3]  = (_Float16)a0.w;
  r[4]  = (_Float16)a1.x; r[5]  = (_Float16)a1.y; r[6]  = (_Float16)a1.z; r[7]  = (_Float16)a1.w;
  r[8]  = (_Float16)a2.x; r[9]  = (_Float16)a2.y; r[10] = (_Float16)a2.z; r[11] = (_Float16)a2.w;
  r[12] = (_Float16)a3.x; r[13] = (_Float16)a3.y; r[14] = (_Float16)a3.z; r[15] = (_Float16)a3.w;
  return r;
}

// ===========================================================================
// Pass 0: bulk f32 -> f16 conversion (one pass; removes all cvt work from the
// GEMM hot loops). 8 elements/thread, b128 in / b128 out.
// ===========================================================================
__global__ void __launch_bounds__(256)
cvt_f32_to_f16(const float* __restrict__ in, unsigned short* __restrict__ out_, int n) {
  _Float16* out = reinterpret_cast<_Float16*>(out_);
  const int i = (blockIdx.x * 256 + threadIdx.x) * 8;
  if (i >= n) return;
  const float4* p = reinterpret_cast<const float4*>(in + i);
  float4 a = p[0], b = p[1];
  v8h o;
  o[0] = (_Float16)a.x; o[1] = (_Float16)a.y; o[2] = (_Float16)a.z; o[3] = (_Float16)a.w;
  o[4] = (_Float16)b.x; o[5] = (_Float16)b.y; o[6] = (_Float16)b.z; o[7] = (_Float16)b.w;
  *reinterpret_cast<v8h*>(out + i) = o;
}

// ===========================================================================
// FAST PATH kernel 1: fused cur = (xh @ W1h^T + b1)/T -> 32-step LIF.
// Wave tile 64x16 (4 WMMA accumulators), block = 4 waves = 64x64 tile.
// `cur` lives only in WMMA accumulators; recurrence is register-resident.
// Spike counts written f32 -> d_out and f16 -> workspace (for GEMM2).
// ===========================================================================
__global__ void __launch_bounds__(128)
snn_fc1_lif_wmma_h(const unsigned short* __restrict__ xh_,
                   const unsigned short* __restrict__ w1h_,
                   const float* __restrict__ b1,
                   const float* __restrict__ tau_mem, const float* __restrict__ tau_syn,
                   const float* __restrict__ v_thresh, const float* __restrict__ v_reset,
                   const float* __restrict__ vm0, const float* __restrict__ is0,
                   const int* __restrict__ time_steps,
                   float* __restrict__ spikes_out,
                   unsigned short* __restrict__ sph_) {
  const _Float16* xh  = reinterpret_cast<const _Float16*>(xh_);
  const _Float16* w1h = reinterpret_cast<const _Float16*>(w1h_);
  _Float16* sph = reinterpret_cast<_Float16*>(sph_);

  const int lane = threadIdx.x & 31;
  const int wave = threadIdx.x >> 5;
  const int n0 = blockIdx.x * 64 + wave * 16;
  const int m0 = blockIdx.y * 64;

  const int aK = (lane >> 4) << 3;
  const int bK = (lane >> 4) << 3;
  const _Float16* aRow = xh  + (size_t)(m0 + (lane & 15)) * INF;
  const _Float16* bRow = w1h + (size_t)(n0 + (lane & 15)) * INF;

  v8f acc[4] = {};
#pragma unroll 2
  for (int k0 = 0; k0 < INF; k0 += 32) {
    __builtin_prefetch(aRow + k0 + 64 + aK, 0, 1);   // speculative; OOB dropped
    __builtin_prefetch(bRow + k0 + 64 + bK, 0, 1);
    v16h bf = load_fragB_h(bRow, k0 + bK);
    v16h a0 = load_fragA_h(aRow,            k0 + aK);
    v16h a1 = load_fragA_h(aRow + 16 * INF, k0 + aK);
    v16h a2 = load_fragA_h(aRow + 32 * INF, k0 + aK);
    v16h a3 = load_fragA_h(aRow + 48 * INF, k0 + aK);
    acc[0] = __builtin_amdgcn_wmma_f32_16x16x32_f16(false, a0, false, bf, (short)0, acc[0], false, false);
    acc[1] = __builtin_amdgcn_wmma_f32_16x16x32_f16(false, a1, false, bf, (short)0, acc[1], false, false);
    acc[2] = __builtin_amdgcn_wmma_f32_16x16x32_f16(false, a2, false, bf, (short)0, acc[2], false, false);
    acc[3] = __builtin_amdgcn_wmma_f32_16x16x32_f16(false, a3, false, bf, (short)0, acc[3], false, false);
  }

  const int   T    = *time_steps;
  const float invT = 1.0f / (float)T;
  const float dS   = 1.0f - 1.0f / *tau_syn;
  const float dM   = 1.0f - 1.0f / *tau_mem;
  const float th   = *v_thresh;
  const float vr   = *v_reset;

  const int h     = n0 + (lane & 15);
  const int rbase = m0 + ((lane >> 4) << 3);
  const float bias = b1[h];

  float cu[32], vv[32], ii[32], cn[32];
#pragma unroll
  for (int t = 0; t < 4; ++t)
#pragma unroll
    for (int r = 0; r < 8; ++r) {
      const int e = t * 8 + r;
      const size_t idx = (size_t)(rbase + t * 16 + r) * HDIM + h;
      cu[e] = (acc[t][r] + bias) * invT;
      vv[e] = vm0[idx];
      ii[e] = is0[idx];
      cn[e] = 0.0f;
    }

  for (int s = 0; s < T; ++s) {
#pragma unroll
    for (int e = 0; e < 32; ++e) {
      ii[e] = ii[e] * dS + cu[e];
      vv[e] = vv[e] * dM + ii[e];
      const bool sp = vv[e] >= th;
      cn[e] += sp ? 1.0f : 0.0f;
      vv[e] = sp ? vr : vv[e];
    }
  }

#pragma unroll
  for (int t = 0; t < 4; ++t)
#pragma unroll
    for (int r = 0; r < 8; ++r) {
      const int e = t * 8 + r;
      const size_t idx = (size_t)(rbase + t * 16 + r) * HDIM + h;
      spikes_out[idx] = cn[e];
      sph[idx] = (_Float16)cn[e];       // exact: counts are 0..32
    }
}

// ===========================================================================
// FAST PATH kernel 2: output = spikes_h @ W2h^T + b2
// ===========================================================================
__global__ void __launch_bounds__(128)
snn_fc2_wmma_h(const unsigned short* __restrict__ sph_,
               const unsigned short* __restrict__ w2h_,
               const float* __restrict__ b2, float* __restrict__ out) {
  const _Float16* sp  = reinterpret_cast<const _Float16*>(sph_);
  const _Float16* w2h = reinterpret_cast<const _Float16*>(w2h_);

  const int lane = threadIdx.x & 31;
  const int wave = threadIdx.x >> 5;
  const int n0 = blockIdx.x * 64 + wave * 16;
  const int m0 = blockIdx.y * 64;

  const int aK = (lane >> 4) << 3;
  const int bK = (lane >> 4) << 3;
  const _Float16* aRow = sp  + (size_t)(m0 + (lane & 15)) * HDIM;
  const _Float16* bRow = w2h + (size_t)(n0 + (lane & 15)) * HDIM;

  v8f acc[4] = {};
#pragma unroll 2
  for (int k0 = 0; k0 < HDIM; k0 += 32) {
    __builtin_prefetch(aRow + k0 + 64 + aK, 0, 1);
    __builtin_prefetch(bRow + k0 + 64 + bK, 0, 1);
    v16h bf = load_fragB_h(bRow, k0 + bK);
    v16h a0 = load_fragA_h(aRow,             k0 + aK);
    v16h a1 = load_fragA_h(aRow + 16 * HDIM, k0 + aK);
    v16h a2 = load_fragA_h(aRow + 32 * HDIM, k0 + aK);
    v16h a3 = load_fragA_h(aRow + 48 * HDIM, k0 + aK);
    acc[0] = __builtin_amdgcn_wmma_f32_16x16x32_f16(false, a0, false, bf, (short)0, acc[0], false, false);
    acc[1] = __builtin_amdgcn_wmma_f32_16x16x32_f16(false, a1, false, bf, (short)0, acc[1], false, false);
    acc[2] = __builtin_amdgcn_wmma_f32_16x16x32_f16(false, a2, false, bf, (short)0, acc[2], false, false);
    acc[3] = __builtin_amdgcn_wmma_f32_16x16x32_f16(false, a3, false, bf, (short)0, acc[3], false, false);
  }

  const int o     = n0 + (lane & 15);
  const int rbase = m0 + ((lane >> 4) << 3);
  const float bias = b2[o];
#pragma unroll
  for (int t = 0; t < 4; ++t)
#pragma unroll
    for (int r = 0; r < 8; ++r)
      out[(size_t)(rbase + t * 16 + r) * ODIM + o] = acc[t][r] + bias;
}

// ===========================================================================
// FALLBACK kernels (inline f32->f16 conversion; used if workspace too small)
// ===========================================================================
__global__ void __launch_bounds__(128)
snn_fc1_lif_wmma(const float* __restrict__ x,  const float* __restrict__ W1,
                 const float* __restrict__ b1,
                 const float* __restrict__ tau_mem, const float* __restrict__ tau_syn,
                 const float* __restrict__ v_thresh, const float* __restrict__ v_reset,
                 const float* __restrict__ vm0, const float* __restrict__ is0,
                 const int* __restrict__ time_steps,
                 float* __restrict__ spikes_out) {
  const int lane = threadIdx.x & 31;
  const int wave = threadIdx.x >> 5;
  const int n0 = blockIdx.x * 64 + wave * 16;
  const int m0 = blockIdx.y * 64;
  const int aK = (lane >> 4) << 3;
  const int bK = (lane >> 4) << 3;
  const float* aRow = x  + (size_t)(m0 + (lane & 15)) * INF;
  const float* bRow = W1 + (size_t)(n0 + (lane & 15)) * INF;

  v8f acc[4] = {};
  for (int k0 = 0; k0 < INF; k0 += 32) {
    v16h bf = load_fragB(bRow, k0 + bK);
    v16h a0 = load_fragA(aRow,            k0 + aK);
    v16h a1 = load_fragA(aRow + 16 * INF, k0 + aK);
    v16h a2 = load_fragA(aRow + 32 * INF, k0 + aK);
    v16h a3 = load_fragA(aRow + 48 * INF, k0 + aK);
    acc[0] = __builtin_amdgcn_wmma_f32_16x16x32_f16(false, a0, false, bf, (short)0, acc[0], false, false);
    acc[1] = __builtin_amdgcn_wmma_f32_16x16x32_f16(false, a1, false, bf, (short)0, acc[1], false, false);
    acc[2] = __builtin_amdgcn_wmma_f32_16x16x32_f16(false, a2, false, bf, (short)0, acc[2], false, false);
    acc[3] = __builtin_amdgcn_wmma_f32_16x16x32_f16(false, a3, false, bf, (short)0, acc[3], false, false);
  }

  const int   T    = *time_steps;
  const float invT = 1.0f / (float)T;
  const float dS   = 1.0f - 1.0f / *tau_syn;
  const float dM   = 1.0f - 1.0f / *tau_mem;
  const float th   = *v_thresh;
  const float vr   = *v_reset;
  const int h     = n0 + (lane & 15);
  const int rbase = m0 + ((lane >> 4) << 3);
  const float bias = b1[h];

  float cu[32], vv[32], ii[32], cn[32];
#pragma unroll
  for (int t = 0; t < 4; ++t)
#pragma unroll
    for (int r = 0; r < 8; ++r) {
      const int e = t * 8 + r;
      const size_t idx = (size_t)(rbase + t * 16 + r) * HDIM + h;
      cu[e] = (acc[t][r] + bias) * invT;
      vv[e] = vm0[idx];
      ii[e] = is0[idx];
      cn[e] = 0.0f;
    }
  for (int s = 0; s < T; ++s) {
#pragma unroll
    for (int e = 0; e < 32; ++e) {
      ii[e] = ii[e] * dS + cu[e];
      vv[e] = vv[e] * dM + ii[e];
      const bool sp = vv[e] >= th;
      cn[e] += sp ? 1.0f : 0.0f;
      vv[e] = sp ? vr : vv[e];
    }
  }
#pragma unroll
  for (int t = 0; t < 4; ++t)
#pragma unroll
    for (int r = 0; r < 8; ++r)
      spikes_out[(size_t)(rbase + t * 16 + r) * HDIM + h] = cn[t * 8 + r];
}

__global__ void __launch_bounds__(128)
snn_fc2_wmma(const float* __restrict__ spikes, const float* __restrict__ W2,
             const float* __restrict__ b2, float* __restrict__ out) {
  const int lane = threadIdx.x & 31;
  const int wave = threadIdx.x >> 5;
  const int n0 = blockIdx.x * 64 + wave * 16;
  const int m0 = blockIdx.y * 64;
  const int aK = (lane >> 4) << 3;
  const int bK = (lane >> 4) << 3;
  const float* aRow = spikes + (size_t)(m0 + (lane & 15)) * HDIM;
  const float* bRow = W2     + (size_t)(n0 + (lane & 15)) * HDIM;

  v8f acc[4] = {};
  for (int k0 = 0; k0 < HDIM; k0 += 32) {
    v16h bf = load_fragB(bRow, k0 + bK);
    v16h a0 = load_fragA(aRow,             k0 + aK);
    v16h a1 = load_fragA(aRow + 16 * HDIM, k0 + aK);
    v16h a2 = load_fragA(aRow + 32 * HDIM, k0 + aK);
    v16h a3 = load_fragA(aRow + 48 * HDIM, k0 + aK);
    acc[0] = __builtin_amdgcn_wmma_f32_16x16x32_f16(false, a0, false, bf, (short)0, acc[0], false, false);
    acc[1] = __builtin_amdgcn_wmma_f32_16x16x32_f16(false, a1, false, bf, (short)0, acc[1], false, false);
    acc[2] = __builtin_amdgcn_wmma_f32_16x16x32_f16(false, a2, false, bf, (short)0, acc[2], false, false);
    acc[3] = __builtin_amdgcn_wmma_f32_16x16x32_f16(false, a3, false, bf, (short)0, acc[3], false, false);
  }
  const int o     = n0 + (lane & 15);
  const int rbase = m0 + ((lane >> 4) << 3);
  const float bias = b2[o];
#pragma unroll
  for (int t = 0; t < 4; ++t)
#pragma unroll
    for (int r = 0; r < 8; ++r)
      out[(size_t)(rbase + t * 16 + r) * ODIM + o] = acc[t][r] + bias;
}

// ===========================================================================
extern "C" void kernel_launch(void* const* d_in, const int* in_sizes, int n_in,
                              void* d_out, int out_size, void* d_ws, size_t ws_size,
                              hipStream_t stream) {
  (void)in_sizes; (void)n_in; (void)out_size;
  const float* x        = (const float*)d_in[0];
  const float* W1       = (const float*)d_in[1];
  const float* b1       = (const float*)d_in[2];
  const float* W2       = (const float*)d_in[3];
  const float* b2       = (const float*)d_in[4];
  const float* tau_mem  = (const float*)d_in[5];
  const float* tau_syn  = (const float*)d_in[6];
  const float* v_thresh = (const float*)d_in[7];
  const float* v_reset  = (const float*)d_in[8];
  const float* vm0      = (const float*)d_in[9];
  const float* is0      = (const float*)d_in[10];
  const int*   tsteps   = (const int*)d_in[11];

  float* out    = (float*)d_out;              // [4096, 256]
  float* spikes = out + (size_t)BSZ * ODIM;   // [4096, 2048]

  const size_t nX  = (size_t)BSZ * INF;       // 4.19M
  const size_t nW1 = (size_t)HDIM * INF;      // 2.10M
  const size_t nW2 = (size_t)ODIM * HDIM;     // 0.52M
  const size_t nSp = (size_t)BSZ * HDIM;      // 8.39M
  const size_t need = (nX + nW1 + nW2 + nSp) * sizeof(unsigned short);  // ~29 MB

  dim3 blk(128, 1, 1);
  dim3 grd1(HDIM / 64, BSZ / 64, 1);
  dim3 grd2(ODIM / 64, BSZ / 64, 1);

  if (ws_size >= need) {
    unsigned short* xh  = (unsigned short*)d_ws;
    unsigned short* w1h = xh + nX;
    unsigned short* w2h = w1h + nW1;
    unsigned short* sph = w2h + nW2;

    cvt_f32_to_f16<<<dim3((unsigned)(nX  / 2048)), dim3(256), 0, stream>>>(x,  xh,  (int)nX);
    cvt_f32_to_f16<<<dim3((unsigned)(nW1 / 2048)), dim3(256), 0, stream>>>(W1, w1h, (int)nW1);
    cvt_f32_to_f16<<<dim3((unsigned)(nW2 / 2048)), dim3(256), 0, stream>>>(W2, w2h, (int)nW2);

    snn_fc1_lif_wmma_h<<<grd1, blk, 0, stream>>>(xh, w1h, b1, tau_mem, tau_syn,
                                                 v_thresh, v_reset, vm0, is0,
                                                 tsteps, spikes, sph);
    snn_fc2_wmma_h<<<grd2, blk, 0, stream>>>(sph, w2h, b2, out);
  } else {
    snn_fc1_lif_wmma<<<grd1, blk, 0, stream>>>(x, W1, b1, tau_mem, tau_syn,
                                               v_thresh, v_reset, vm0, is0,
                                               tsteps, spikes);
    snn_fc2_wmma<<<grd2, blk, 0, stream>>>(spikes, W2, b2, out);
  }
}
